// SPRGraphNet_88648124990151
// MI455X (gfx1250) — matrix operand order repaired and checked by reading, not compiled, not run on hardware
//
#include <hip/hip_runtime.h>

#define N_NODES  50000
#define N_EDGES  1600000
#define N_GRAPHS 512
#define EMB      32
#define HID      64
#define NCLS     2

typedef _Float16 v16h __attribute__((ext_vector_type(16)));
typedef float    v8f  __attribute__((ext_vector_type(8)));

__global__ void zero_f32(float* __restrict__ p, int n) {
  int t = blockIdx.x * blockDim.x + threadIdx.x;
  if (t < n) p[t] = 0.0f;
}

// x[i, 0:32] = embed[ids[i], 0:32], float4-vectorized (8 chunks/node)
__global__ void embed_gather(const int* __restrict__ ids,
                             const float* __restrict__ emb,
                             float* __restrict__ x) {
  int t = blockIdx.x * blockDim.x + threadIdx.x;
  if (t >= N_NODES * (EMB / 4)) return;
  int i = t >> 3;
  int c = (t & 7) << 2;
  *(float4*)(x + (size_t)i * EMB + c) =
      *(const float4*)(emb + (size_t)ids[i] * EMB + c);
}

__global__ void deg_count(const int* __restrict__ dst, float* __restrict__ deg) {
  int e = blockIdx.x * blockDim.x + threadIdx.x;
  if (e < N_EDGES) atomicAdd(&deg[dst[e]], 1.0f);
}

// deg[i] -> 1 / max(deg[i], 1)   (hoists the f32 divide out of the GEMM kernel)
__global__ void finalize_invdeg(float* __restrict__ deg, int n) {
  int i = blockIdx.x * blockDim.x + threadIdx.x;
  if (i < n) deg[i] = 1.0f / fmaxf(deg[i], 1.0f);
}

__global__ void graph_count(const int* __restrict__ batch, float* __restrict__ gcnt) {
  int i = blockIdx.x * blockDim.x + threadIdx.x;
  if (i < N_NODES) atomicAdd(&gcnt[batch[i]], 1.0f);
}

// wT[col*K + k] = (f16) w[k*HID + col]  — transposed f16 weights for vector B loads
template <int K>
__global__ void prep_wT(const float* __restrict__ w, _Float16* __restrict__ wT) {
  int t = blockIdx.x * blockDim.x + threadIdx.x;
  if (t >= HID * K) return;
  int col = t / K;
  int k   = t - col * K;
  wT[t] = (_Float16)w[k * HID + col];
}

// 4 threads per edge, each handles NFEAT/4 contiguous floats (float4 loads +
// f32 atomics into the L2-resident message buffer).
template <int NFEAT>
__global__ void scatter_edges(const int* __restrict__ src, const int* __restrict__ dst,
                              const float* __restrict__ x, float* __restrict__ msg) {
  int t = blockIdx.x * blockDim.x + threadIdx.x;
  int e = t >> 2;
  if (e >= N_EDGES) return;
  const int q = (t & 3) * (NFEAT / 4);
  const float* xs = x + (size_t)src[e] * NFEAT + q;
  float* p = msg + (size_t)dst[e] * NFEAT + q;
#pragma unroll
  for (int c = 0; c < NFEAT / 4; c += 4) {
    const float4 v = *(const float4*)(xs + c);
    atomicAdd(p + c + 0, v.x);
    atomicAdd(p + c + 1, v.y);
    atomicAdd(p + c + 2, v.z);
    atomicAdd(p + c + 3, v.w);
  }
}

// out[node, 0..63] = relu( (msg*invdeg) @ wl + self @ wr + bias ), K = input dim
// One wave per 16-node tile; 4 column tiles of 16; f16 WMMA, f32 accumulate.
template <int K>
__global__ void node_update(const float* __restrict__ msg, const float* __restrict__ invdeg,
                            const float* __restrict__ self,
                            const _Float16* __restrict__ wlT, const _Float16* __restrict__ wrT,
                            const float* __restrict__ bias, float* __restrict__ out) {
  constexpr int KV = K / 32;
  const int lane = threadIdx.x & 31;
  const int wave = threadIdx.x >> 5;
  const int tile = blockIdx.x * (blockDim.x >> 5) + wave;
  if (tile * 16 >= N_NODES) return;          // uniform per wave, EXEC stays full
  const int rc   = lane & 15;                // A-row / B,D-column within tile
  const int half = lane >> 4;
  const int node = tile * 16 + rc;
  const float invd = invdeg[node];

  // A matrices (16x32 f16 per chunk): lane holds row `rc`;
  // elements 0..15 -> K = base..base+15,
  // base = kv*32 + half*16. Contiguous in memory -> float4 loads.
  v16h a_m[KV], a_s[KV];
#pragma unroll
  for (int kv = 0; kv < KV; ++kv) {
    const float* pm = msg  + (size_t)node * K + kv * 32 + half * 16;
    const float* ps = self + (size_t)node * K + kv * 32 + half * 16;
    float tm[16], ts[16];
    *(float4*)(tm + 0)  = *(const float4*)(pm + 0);
    *(float4*)(tm + 4)  = *(const float4*)(pm + 4);
    *(float4*)(tm + 8)  = *(const float4*)(pm + 8);
    *(float4*)(tm + 12) = *(const float4*)(pm + 12);
    *(float4*)(ts + 0)  = *(const float4*)(ps + 0);
    *(float4*)(ts + 4)  = *(const float4*)(ps + 4);
    *(float4*)(ts + 8)  = *(const float4*)(ps + 8);
    *(float4*)(ts + 12) = *(const float4*)(ps + 12);
#pragma unroll
    for (int e = 0; e < 16; ++e) {
      a_m[kv][e] = (_Float16)(tm[e] * invd);
      a_s[kv][e] = (_Float16)(ts[e]);
    }
  }

#pragma unroll
  for (int t = 0; t < HID / 16; ++t) {
    const int col = t * 16 + rc;
    v8f acc;
    const float bv = bias[col];
#pragma unroll
    for (int r = 0; r < 8; ++r) acc[r] = bv;
#pragma unroll
    for (int kv = 0; kv < KV; ++kv) {
      // B (32x16 per chunk): lane holds column `col`; element e -> K = kv*32 + half*16 + e.
      // Pre-transposed f16 weights make this one 32B vector load per operand.
      const v16h bl = *(const v16h*)(wlT + (size_t)col * K + kv * 32 + half * 16);
      const v16h br = *(const v16h*)(wrT + (size_t)col * K + kv * 32 + half * 16);
      acc = __builtin_amdgcn_wmma_f32_16x16x32_f16(false, a_s[kv], false, br,
                                                   (short)0, acc, false, false);
      acc = __builtin_amdgcn_wmma_f32_16x16x32_f16(false, a_m[kv], false, bl,
                                                   (short)0, acc, false, false);
    }
    // D layout: lane holds column `col`; VGPR r -> row half*8 + r
#pragma unroll
    for (int r = 0; r < 8; ++r) {
      const int m = half * 8 + r;
      out[(size_t)(tile * 16 + m) * HID + col] = fmaxf(acc[r], 0.0f);
    }
  }
}

__global__ void pool_scatter(const int* __restrict__ batch,
                             const float* __restrict__ h,
                             float* __restrict__ gsum) {
  int t = blockIdx.x * blockDim.x + threadIdx.x;
  if (t >= N_NODES * (HID / 4)) return;
  int i = t >> 4;
  int c = (t & 15) << 2;
  const float4 v = *reinterpret_cast<const float4*>(h + (size_t)i * HID + c);
  float* p = gsum + (size_t)batch[i] * HID + c;
  atomicAdd(p + 0, v.x);
  atomicAdd(p + 1, v.y);
  atomicAdd(p + 2, v.z);
  atomicAdd(p + 3, v.w);
}

__global__ void out_head(const float* __restrict__ gsum, const float* __restrict__ gcnt,
                         const float* __restrict__ w, const float* __restrict__ b,
                         float* __restrict__ out) {
  int t = blockIdx.x * blockDim.x + threadIdx.x;
  if (t >= N_GRAPHS * NCLS) return;
  int g = t / NCLS, c = t % NCLS;
  float invc = 1.0f / fmaxf(gcnt[g], 1.0f);
  float s = b[c];
#pragma unroll
  for (int f = 0; f < HID; ++f) s += gsum[g * HID + f] * invc * w[f * NCLS + c];
  out[t] = s;
}

extern "C" void kernel_launch(void* const* d_in, const int* in_sizes, int n_in,
                              void* d_out, int out_size, void* d_ws, size_t ws_size,
                              hipStream_t stream) {
  const int*   x_ids = (const int*)d_in[0];
  const int*   ei    = (const int*)d_in[1];
  const int*   src   = ei;
  const int*   dst   = ei + N_EDGES;
  const int*   batch = (const int*)d_in[2];
  const float* embed = (const float*)d_in[3];
  const float* w1_l  = (const float*)d_in[4];
  const float* b1    = (const float*)d_in[5];
  const float* w1_r  = (const float*)d_in[6];
  const float* w2_l  = (const float*)d_in[7];
  const float* b2    = (const float*)d_in[8];
  const float* w2_r  = (const float*)d_in[9];
  const float* w_out = (const float*)d_in[10];
  const float* b_out = (const float*)d_in[11];
  float* out = (float*)d_out;

  // Workspace layout (f32): X | DEG | MSG | H1 | H2 | GSUM | GCNT | f16 weights
  float* X    = (float*)d_ws;
  float* DEG  = X    + (size_t)N_NODES * EMB;
  float* MSG  = DEG  + (size_t)N_NODES;
  float* H1   = MSG  + (size_t)N_NODES * HID;
  float* H2   = H1   + (size_t)N_NODES * HID;
  float* GSUM = H2   + (size_t)N_NODES * HID;
  float* GCNT = GSUM + (size_t)N_GRAPHS * HID;
  _Float16* WT1L = (_Float16*)(GCNT + N_GRAPHS);   // [HID][EMB]
  _Float16* WT1R = WT1L + (size_t)HID * EMB;       // [HID][EMB]
  _Float16* WT2L = WT1R + (size_t)HID * EMB;       // [HID][HID]
  _Float16* WT2R = WT2L + (size_t)HID * HID;       // [HID][HID]

  auto cdiv = [](int a, int b) { return (a + b - 1) / b; };
  const int B = 256;
  const int TILES = N_NODES / 16;                 // 3125 exactly
  const int UPD_BLOCKS = cdiv(TILES, 4);          // 4 waves / block

  // zero DEG..MSG (contiguous) and GSUM..GCNT (contiguous)
  zero_f32<<<cdiv(N_NODES + N_NODES * HID, B), B, 0, stream>>>(DEG, N_NODES + N_NODES * HID);
  zero_f32<<<cdiv(N_GRAPHS * HID + N_GRAPHS, B), B, 0, stream>>>(GSUM, N_GRAPHS * HID + N_GRAPHS);

  // one-shot f16 transposed weight prep (tiny)
  prep_wT<EMB><<<cdiv(HID * EMB, B), B, 0, stream>>>(w1_l, WT1L);
  prep_wT<EMB><<<cdiv(HID * EMB, B), B, 0, stream>>>(w1_r, WT1R);
  prep_wT<HID><<<cdiv(HID * HID, B), B, 0, stream>>>(w2_l, WT2L);
  prep_wT<HID><<<cdiv(HID * HID, B), B, 0, stream>>>(w2_r, WT2R);

  embed_gather<<<cdiv(N_NODES * (EMB / 4), B), B, 0, stream>>>(x_ids, embed, X);
  deg_count<<<cdiv(N_EDGES, B), B, 0, stream>>>(dst, DEG);
  finalize_invdeg<<<cdiv(N_NODES, B), B, 0, stream>>>(DEG, N_NODES);
  graph_count<<<cdiv(N_NODES, B), B, 0, stream>>>(batch, GCNT);

  // Layer 1: aggregate EMB=32-wide messages, then WMMA node update -> H1
  scatter_edges<EMB><<<cdiv(N_EDGES * 4, B), B, 0, stream>>>(src, dst, X, MSG);
  node_update<EMB><<<UPD_BLOCKS, 128, 0, stream>>>(MSG, DEG, X, WT1L, WT1R, b1, H1);

  // Layer 2: re-zero MSG, aggregate HID=64-wide, WMMA node update -> H2
  zero_f32<<<cdiv(N_NODES * HID, B), B, 0, stream>>>(MSG, N_NODES * HID);
  scatter_edges<HID><<<cdiv(N_EDGES * 4, B), B, 0, stream>>>(src, dst, H1, MSG);
  node_update<HID><<<UPD_BLOCKS, 128, 0, stream>>>(MSG, DEG, H1, WT2L, WT2R, b2, H2);

  // Global mean pool + linear head
  pool_scatter<<<cdiv(N_NODES * (HID / 4), B), B, 0, stream>>>(batch, H2, GSUM);
  out_head<<<cdiv(N_GRAPHS * NCLS, B), B, 0, stream>>>(GSUM, GCNT, w_out, b_out, out);
}
